// GINNet_74440373175008
// MI455X (gfx1250) — matrix operand order repaired and checked by reading, not compiled, not run on hardware
//
#include <hip/hip_runtime.h>

// ---------------------------------------------------------------------------
// GIN forward for MI455X (gfx1250, wave32).
// Bottleneck: edge gather/scatter (~5.6 GB, L2-resident on the 192MB L2) ->
// guaranteed hw global_atomic_add_f32 via inline asm (L2 atomic units; WGP$ is
// read-only so device-coherent). GEMMs (~19.4 GFLOP) via
// v_wmma_f32_16x16x32_bf16, 2 row-tiles per wave, BN column stats fused into
// the GEMM epilogue.
// ---------------------------------------------------------------------------

typedef __bf16 bf16;
typedef __attribute__((ext_vector_type(16))) bf16  v16bf;
typedef __attribute__((ext_vector_type(8)))  bf16  v8bf;
typedef __attribute__((ext_vector_type(8)))  float v8f;

#define HID      110
#define HPAD     112     // f32 row stride (16B-aligned rows)
#define KPAD     128     // bf16 GEMM K padding (4 x K=32 WMMA steps)
#define BN_EPS   1e-5f

// Small-count atomics (BN stats): HIP builtin is fine.
__device__ __forceinline__ void atomAddF(float* p, float v) {
    __hip_atomic_fetch_add(p, v, __ATOMIC_RELAXED, __HIP_MEMORY_SCOPE_AGENT);
}

// Hot-path atomic: guaranteed single hw instruction, executes in L2 atomic
// units at device scope (no CAS-loop fallback risk).
__device__ __forceinline__ void atomAddFHW(float* p, float v) {
    asm volatile("global_atomic_add_f32 %0, %1, off scope:SCOPE_DEV"
                 :: "v"(p), "v"(v) : "memory");
}

// h[n][c] = emb[x[n]][c], pad cols 110..111 with 0
__global__ void embed_kernel(const int* __restrict__ x,
                             const float* __restrict__ emb,
                             float* __restrict__ h, int n_nodes) {
    int idx = blockIdx.x * 256 + threadIdx.x;
    int total = n_nodes * HPAD;
    if (idx >= total) return;
    int n = idx / HPAD, c = idx - n * HPAD;
    h[idx] = (c < HID) ? emb[x[n] * HID + c] : 0.0f;
}

// scatter: agg[dst] += relu(h[src]); one thread = one (edge, 4-channel group).
// 28 consecutive threads share an edge -> index loads coalesce/broadcast,
// h-row reads are contiguous 448B bursts. Zero messages are skipped (relu
// kills ~half), halving atomic traffic.
__global__ void edge_scatter_kernel(const float* __restrict__ h,
                                    const int* __restrict__ ei,
                                    float* __restrict__ agg, int n_edges) {
    int idx = blockIdx.x * 256 + threadIdx.x;
    const int GPE = HPAD / 4;                    // 28 groups per edge
    if (idx >= n_edges * GPE) return;
    int e = idx / GPE, g = idx - e * GPE;
    int src = ei[e];
    int dst = ei[n_edges + e];
    const float4 v = *(const float4*)(h + (size_t)src * HPAD + g * 4);
    float* ap = agg + (size_t)dst * HPAD + g * 4;
    float v0 = fmaxf(v.x, 0.f), v1 = fmaxf(v.y, 0.f);
    float v2 = fmaxf(v.z, 0.f), v3 = fmaxf(v.w, 0.f);
    if (v0 != 0.f) atomAddFHW(ap + 0, v0);
    if (v1 != 0.f) atomAddFHW(ap + 1, v1);
    if (v2 != 0.f) atomAddFHW(ap + 2, v2);
    if (v3 != 0.f) atomAddFHW(ap + 3, v3);
}

// zb[n][k] = bf16((1+eps)*h + agg), zero-padded to KPAD
__global__ void zb_kernel(const float* __restrict__ h,
                          const float* __restrict__ agg,
                          const float* __restrict__ eps_p, int layer,
                          bf16* __restrict__ zb, int n_nodes) {
    int idx = blockIdx.x * 256 + threadIdx.x;
    if (idx >= n_nodes * KPAD) return;
    int n = idx >> 7, c = idx & (KPAD - 1);
    float v = 0.0f;
    if (c < HID) {
        float ep = 1.0f + eps_p[layer];
        v = ep * h[(size_t)n * HPAD + c] + agg[(size_t)n * HPAD + c];
    }
    zb[idx] = (bf16)v;
}

// Wt[n][k] = bf16(W[k][n]), zero-padded [HPAD][KPAD]
__global__ void transpose_w_kernel(const float* __restrict__ W,
                                   bf16* __restrict__ Wt) {
    int n = blockIdx.x;              // 0..111
    int k = threadIdx.x;             // 0..127
    float v = (n < HID && k < HID) ? W[k * HID + n] : 0.0f;
    Wt[n * KPAD + k] = (bf16)v;
}

// out[M][112] = A[M][128](bf16) x Bt[112][128]^T + bias ; col sum/sumsq atomics.
// Block: 224 thr = 7 waves; wave w -> cols [16w,16w+16); block -> rows
// [32b, 32b+32): two 16-row tiles per wave, B fragment reused across both.
__global__ __launch_bounds__(224)
void gemm_bf16_kernel(const bf16* __restrict__ A,
                      const bf16* __restrict__ Bt,
                      const float* __restrict__ bias,
                      float* __restrict__ out,
                      float* __restrict__ colsum,
                      float* __restrict__ colsumsq) {
    const int wave = threadIdx.x >> 5;
    const int lane = threadIdx.x & 31;
    const int l15  = lane & 15;
    const int hi   = lane >> 4;                 // 0 or 1
    const int rowbase = blockIdx.x * 32;
    const int col     = wave * 16 + l15;        // 0..111

    float bv = (col < HID) ? bias[col] : 0.0f;
    v8f acc0, acc1;
#pragma unroll
    for (int j = 0; j < 8; ++j) { acc0[j] = bv; acc1[j] = bv; }

    const bf16* arow0 = A  + (size_t)(rowbase + l15) * KPAD;
    const bf16* arow1 = arow0 + (size_t)16 * KPAD;
    const bf16* brow  = Bt + (size_t)col * KPAD;

#pragma unroll
    for (int kb = 0; kb < KPAD; kb += 32) {
        union { v16bf v; v8bf h[2]; } a0, a1, bfm;
        // B 32x16 (rows of Bt = weight columns): element j<8 -> K=kb+hi*8+j ; j>=8 -> K=kb+16+hi*8+(j-8)
        bfm.h[0] = *(const v8bf*)(brow + kb + hi * 8);
        bfm.h[1] = *(const v8bf*)(brow + kb + 16 + hi * 8);
        // A 16x32: element j<8 -> K=kb+hi*8+j ; j>=8 -> K=kb+16+hi*8+(j-8)
        a0.h[0]  = *(const v8bf*)(arow0 + kb + hi * 8);
        a0.h[1]  = *(const v8bf*)(arow0 + kb + 16 + hi * 8);
        a1.h[0]  = *(const v8bf*)(arow1 + kb + hi * 8);
        a1.h[1]  = *(const v8bf*)(arow1 + kb + 16 + hi * 8);
        acc0 = __builtin_amdgcn_wmma_f32_16x16x32_bf16(
            false, a0.v, false, bfm.v, (short)0, acc0, false, false);
        acc1 = __builtin_amdgcn_wmma_f32_16x16x32_bf16(
            false, a1.v, false, bfm.v, (short)0, acc1, false, false);
    }

    float s = 0.f, s2 = 0.f;
#pragma unroll
    for (int j = 0; j < 8; ++j) {
        int m = rowbase + j + hi * 8;           // D: VGPR j -> row j (+8 for hi lanes)
        float v = acc0[j];
        out[(size_t)m * HPAD + col] = v;
        s += v; s2 += v * v;
    }
#pragma unroll
    for (int j = 0; j < 8; ++j) {
        int m = rowbase + 16 + j + hi * 8;
        float v = acc1[j];
        out[(size_t)m * HPAD + col] = v;
        s += v; s2 += v * v;
    }
    if (col < HID) {
        atomAddF(&colsum[col],   s);
        atomAddF(&colsumsq[col], s2);
    }
}

// fold BN into per-column affine: a = g*rsqrt(var+eps), c = beta - mean*a
__global__ void bnparam_kernel(const float* __restrict__ colsum,
                               const float* __restrict__ colsumsq,
                               const float* __restrict__ g,
                               const float* __restrict__ b,
                               float* __restrict__ abuf,
                               float* __restrict__ cbuf, int n_nodes) {
    int c = threadIdx.x;
    if (c >= HID) return;
    float inv_n = 1.0f / (float)n_nodes;
    float mean = colsum[c] * inv_n;
    float var  = colsumsq[c] * inv_n - mean * mean;
    float inv  = rsqrtf(var + BN_EPS);
    float a    = g[c] * inv;
    abuf[c] = a;
    cbuf[c] = b[c] - mean * a;
}

// ub[n][k] = bf16(relu(t*a + c)), zero-padded to KPAD
__global__ void ub_kernel(const float* __restrict__ t,
                          const float* __restrict__ abuf,
                          const float* __restrict__ cbuf,
                          bf16* __restrict__ ub, int n_nodes) {
    int idx = blockIdx.x * 256 + threadIdx.x;
    if (idx >= n_nodes * KPAD) return;
    int n = idx >> 7, c = idx & (KPAD - 1);
    float v = 0.0f;
    if (c < HID)
        v = fmaxf(t[(size_t)n * HPAD + c] * abuf[c] + cbuf[c], 0.0f);
    ub[idx] = (bf16)v;
}

// h += relu(z2*a + c)   (pad cols untouched, stay 0)
__global__ void update_h_kernel(const float* __restrict__ z2,
                                const float* __restrict__ abuf,
                                const float* __restrict__ cbuf,
                                float* __restrict__ h, int n_nodes) {
    int idx = blockIdx.x * 256 + threadIdx.x;
    if (idx >= n_nodes * HPAD) return;
    int c = idx % HPAD;
    if (c >= HID) return;
    h[idx] += fmaxf(z2[idx] * abuf[c] + cbuf[c], 0.0f);
}

// score[n][0..1] (+)= h[n] @ Wp + bp   (Wp/bp loads are wave-uniform -> SMEM)
__global__ void score_kernel(const float* __restrict__ h,
                             const float* __restrict__ Wp,   // [110][2]
                             const float* __restrict__ bp,   // [2]
                             float* __restrict__ out, int n_nodes, int accumulate) {
    int n = blockIdx.x * 256 + threadIdx.x;
    if (n >= n_nodes) return;
    float a0 = bp[0], a1 = bp[1];
    const float* hr = h + (size_t)n * HPAD;
#pragma unroll 2
    for (int k = 0; k < HID; ++k) {
        float hv = hr[k];
        a0 += hv * Wp[k * 2 + 0];
        a1 += hv * Wp[k * 2 + 1];
    }
    if (accumulate) { out[n * 2] += a0; out[n * 2 + 1] += a1; }
    else            { out[n * 2]  = a0; out[n * 2 + 1]  = a1; }
}

extern "C" void kernel_launch(void* const* d_in, const int* in_sizes, int n_in,
                              void* d_out, int out_size, void* d_ws, size_t ws_size,
                              hipStream_t stream) {
    const int*   x     = (const int*)d_in[0];
    const int*   ei    = (const int*)d_in[1];
    const float* emb   = (const float*)d_in[2];
    const float* W1    = (const float*)d_in[3];
    const float* b1    = (const float*)d_in[4];
    const float* W2    = (const float*)d_in[5];
    const float* b2    = (const float*)d_in[6];
    const float* g_mlp = (const float*)d_in[7];
    const float* be_mlp= (const float*)d_in[8];
    const float* eps_p = (const float*)d_in[9];
    const float* g_nd  = (const float*)d_in[10];
    const float* be_nd = (const float*)d_in[11];
    const float* Wp    = (const float*)d_in[12];
    const float* bp    = (const float*)d_in[13];
    float* out = (float*)d_out;

    const int N = in_sizes[0];          // 100000
    const int E = in_sizes[1] / 2;      // 1600000

    // ---- workspace carve-up (aligned to 256B) ----
    char* p = (char*)d_ws;
    auto take = [&](size_t bytes) { char* r = p; p += (bytes + 255) & ~(size_t)255; return r; };
    float* h    = (float*)take((size_t)N * HPAD * sizeof(float));   // 44.8 MB
    float* agg  = (float*)take((size_t)N * HPAD * sizeof(float));   // 44.8 MB, reused as t / z2
    bf16*  zb   = (bf16*) take((size_t)N * KPAD * sizeof(bf16));    // 25.6 MB, reused as ub
    bf16*  W1t  = (bf16*) take((size_t)HPAD * KPAD * sizeof(bf16));
    bf16*  W2t  = (bf16*) take((size_t)HPAD * KPAD * sizeof(bf16));
    float* stats= (float*)take(4 * HPAD * sizeof(float));           // sum, sumsq, a, c
    float* csum = stats, *csq = stats + HPAD, *abuf = stats + 2 * HPAD, *cbuf = stats + 3 * HPAD;
    float* t  = agg;    // alias: agg consumed before t written
    float* z2 = agg;    // alias: t consumed before z2 written
    bf16*  ub = zb;     // alias: zb consumed before ub written

    const int TPB = 256;
    const int grid_np   = (N * HPAD + TPB - 1) / TPB;
    const int grid_nk   = (N * KPAD + TPB - 1) / TPB;
    const int grid_edge = (E * (HPAD / 4) + TPB - 1) / TPB;
    const int grid_gemm = N / 32;       // 3125 (N % 32 == 0)
    const int grid_node = (N + TPB - 1) / TPB;

    // hidden[0]
    embed_kernel<<<grid_np, TPB, 0, stream>>>(x, emb, h, N);
    score_kernel<<<grid_node, TPB, 0, stream>>>(h, Wp, bp, out, N, 0);

    for (int i = 0; i < 4; ++i) {
        // aggregation
        hipMemsetAsync(agg, 0, (size_t)N * HPAD * sizeof(float), stream);
        edge_scatter_kernel<<<grid_edge, TPB, 0, stream>>>(h, ei, agg, E);
        zb_kernel<<<grid_nk, TPB, 0, stream>>>(h, agg, eps_p, i, zb, N);

        // weights -> bf16, column-major, padded
        transpose_w_kernel<<<HPAD, KPAD, 0, stream>>>(W1 + (size_t)i * HID * HID, W1t);
        transpose_w_kernel<<<HPAD, KPAD, 0, stream>>>(W2 + (size_t)i * HID * HID, W2t);

        // t = z @ W1 + b1  (+ BN stats)
        hipMemsetAsync(stats, 0, 2 * HPAD * sizeof(float), stream);
        gemm_bf16_kernel<<<grid_gemm, 224, 0, stream>>>(zb, W1t, b1 + i * HID, t, csum, csq);
        bnparam_kernel<<<1, 128, 0, stream>>>(csum, csq, g_mlp + i * HID, be_mlp + i * HID,
                                              abuf, cbuf, N);
        ub_kernel<<<grid_nk, TPB, 0, stream>>>(t, abuf, cbuf, ub, N);

        // z2 = u @ W2 + b2  (+ BN stats)
        hipMemsetAsync(stats, 0, 2 * HPAD * sizeof(float), stream);
        gemm_bf16_kernel<<<grid_gemm, 224, 0, stream>>>(ub, W2t, b2 + i * HID, z2, csum, csq);
        bnparam_kernel<<<1, 128, 0, stream>>>(csum, csq, g_nd + i * HID, be_nd + i * HID,
                                              abuf, cbuf, N);

        // residual update + score contribution
        update_h_kernel<<<grid_np, TPB, 0, stream>>>(z2, abuf, cbuf, h, N);
        score_kernel<<<grid_node, TPB, 0, stream>>>(h, Wp + (size_t)(i + 1) * HID * 2,
                                                    bp + (i + 1) * 2, out, N, 1);
    }
}
